// Cluster_66228395704512
// MI455X (gfx1250) — matrix operand order repaired and checked by reading, not compiled, not run on hardware
//
#include <hip/hip_runtime.h>
#include <hip/hip_bf16.h>
#include <math.h>
#include <stdint.h>

#if defined(__has_builtin)
#  if __has_builtin(__builtin_amdgcn_global_load_async_to_lds_b128)
#    define ASYNC_LDS_OK 1
#  endif
#endif

typedef __bf16 bf16_t;
typedef __attribute__((ext_vector_type(16))) __bf16 v16bf;
typedef __attribute__((ext_vector_type(8)))  float  v8f;
typedef int v4i_vs __attribute__((__vector_size__(16)));

#define WMMA_BF16(a, b, c) \
  __builtin_amdgcn_wmma_f32_16x16x32_bf16(false, (a), false, (b), (short)0, (c), false, false)

#ifdef ASYNC_LDS_OK
// 16B global -> LDS async copy (ASYNCcnt-tracked, no VGPR round trip).
// Builtin takes (v4i global*, v4i LDS*, imm offset, imm cpol).
// Flat->LDS conversion by 32-bit truncation matches the ISA aperture rule
// flat->global is numerically identity.
__device__ __forceinline__ void async_copy16(const void* g, void* l) {
  __builtin_amdgcn_global_load_async_to_lds_b128(
      (__attribute__((address_space(1))) v4i_vs*)(uintptr_t)g,
      (__attribute__((address_space(3))) v4i_vs*)(unsigned int)(uintptr_t)l,
      0, 0);
}
__device__ __forceinline__ void wait_async0() {
#  if __has_builtin(__builtin_amdgcn_s_wait_asynccnt)
  __builtin_amdgcn_s_wait_asynccnt(0);
#  else
  asm volatile("s_wait_asynccnt 0x0" ::: "memory");
#  endif
}
#endif

// --- A operand: 16x32 bf16 tile, rows (M) striped per lane, documented K interleave ---
// lane L: M = L%16, half h = L/16.  element e: K = (e/8)*16 + h*8 + ((e/2)%4)*2 + (e&1)
__device__ __forceinline__ v16bf load_A_tile(const bf16_t* base, int row_stride,
                                             int m_base, int k_base, int lane) {
  const int m = m_base + (lane & 15);
  const int h = lane >> 4;
  const bf16_t* rp = base + m * row_stride + k_base + (h << 3);
  v16bf a;
#pragma unroll
  for (int e = 0; e < 16; ++e) {
    int k = ((e >> 3) << 4) + (((e >> 1) & 3) << 1) + (e & 1);
    a[e] = rp[k];
  }
  return a;
}

// --- B operand: 32x16 bf16 (KxN), N = lane%16, K = (e/8)*16 + (lane/16)*8 + e%8 ---
__device__ __forceinline__ v16bf load_B_tile(const bf16_t* base, int row_stride,
                                             int n_base, int k_base, int lane) {
  const bf16_t* rp = base + (n_base + (lane & 15)) * row_stride + k_base + ((lane >> 4) << 3);
  v16bf b;
#pragma unroll
  for (int e = 0; e < 16; ++e) b[e] = rp[((e >> 3) << 4) + (e & 7)];
  return b;
}

// B=32, DIM=128, W=H=64, HEADS=8, HD=32, INNER=256, quadrants 2x2 (32x32), centers 2x2.
// pg workspace is PIXEL-MAJOR: pg[(b*4096 + pixel)*256 + ic], bf16.
// Kernel 1: one workgroup per (batch, head, quadrant) region. Fuses conv1x1 (Wf & Wv),
// center pooling, cosine/sigmoid/top-1 assignment, aggregation and dispatch.
__global__ __launch_bounds__(256) void region_cluster_kernel(
    const float* __restrict__ p,  const float* __restrict__ Wf, const float* __restrict__ bfv,
    const float* __restrict__ Wv, const float* __restrict__ bvv,
    const float* __restrict__ alpha_p, const float* __restrict__ beta_p,
    bf16_t* __restrict__ pg)
{
  __shared__ __align__(16) bf16_t sWf[32 * 128];   //  8 KB head slice of Wf (bf16)
  __shared__ __align__(16) bf16_t sWv[32 * 128];   //  8 KB
#ifdef ASYNC_LDS_OK
  __shared__ __align__(16) float  sPf[128 * 64];   // 32 KB async landing: [k][pix] f32
#endif
  __shared__ __align__(16) bf16_t sP [64 * 128];   // 16 KB GEMM staging: [pix][k] bf16
  __shared__ __align__(16) bf16_t sPs[1024 * 32];  // 64 KB p_s for region, [n][c]
  __shared__ __align__(16) bf16_t sPv[1024 * 32];  // 64 KB p_v for region, [n][c]
  __shared__ float  sCentS[128];
  __shared__ float  sCentV[128];
  __shared__ float  sAgg[128];
  __shared__ float  sDen[4];
  __shared__ float  sNrm[4];
  __shared__ float  sS[1024];
  __shared__ unsigned char sIdx[1024];

  const int t    = threadIdx.x;
  const int lane = t & 31;
  const int wvid = t >> 5;
  const int r    = blockIdx.x;         // ((b*8 + e)*2 + f1)*2 + f2
  const int b    = r >> 5;
  const int e    = (r >> 2) & 7;
  const int f1   = (r >> 1) & 1;
  const int f2   = r & 1;

  const float alpha = alpha_p[0];
  const float beta  = beta_p[0];

  { // weights -> LDS as bf16 (head slice: rows e*32 .. e*32+31, 128 K each)
    const float* wf  = Wf + (e * 32) * 128;
    const float* wvp = Wv + (e * 32) * 128;
#pragma unroll
    for (int i = 0; i < 16; ++i) {
      int idx = i * 256 + t;
      sWf[idx] = (bf16_t)wf[idx];
      sWv[idx] = (bf16_t)wvp[idx];
    }
  }
  __syncthreads();

  const float* bfh = bfv + e * 32;
  const float* bvh = bvv + e * 32;

  // ---- fused conv1x1 (s & v) over the region, 16 chunks of 64 pixels ----
  for (int cp = 0; cp < 16; ++cp) {
#ifdef ASYNC_LDS_OK
    // 2048 x 16B async transfers: contiguous global f32 rows -> sPf[k][pix]
#pragma unroll
    for (int i = 0; i < 8; ++i) {
      int s  = i * 256 + t;
      int k  = s >> 4;
      int r2 = (s >> 3) & 1;             // row within chunk (x = cp*2 + r2)
      int ys = s & 7;                    // 4-float segment along Y
      const float* src = p + ((size_t)((b * 128 + k) * 64 + f1 * 32 + cp * 2 + r2)) * 64
                           + f2 * 32 + ys * 4;
      async_copy16(src, &sPf[k * 64 + r2 * 32 + ys * 4]);
    }
    wait_async0();
    __syncthreads();
    // convert pass: f32 [k][pix] -> bf16 [pix][k]
#pragma unroll
    for (int i = 0; i < 32; ++i) {
      int idx = i * 256 + t;
      int pix = idx & 63;
      int k   = idx >> 6;
      sP[pix * 128 + k] = (bf16_t)sPf[k * 64 + pix];
    }
#else
#pragma unroll
    for (int i = 0; i < 32; ++i) {
      int idx = i * 256 + t;
      int pix = idx & 63;
      int k   = idx >> 6;
      int n   = cp * 64 + pix;
      int X   = f1 * 32 + (n >> 5);
      int Y   = f2 * 32 + (n & 31);
      sP[pix * 128 + k] = (bf16_t)p[((b * 128 + k) * 64 + X) * 64 + Y];
    }
#endif
    __syncthreads();

    // 16 output tiles per chunk (4 pixtiles x 2 octiles x {s,v}); 2 per wave
#pragma unroll
    for (int j = 0; j < 2; ++j) {
      int tt = wvid * 2 + j;
      int pixtile = tt & 3;
      int octile  = (tt >> 2) & 1;
      int mat     = tt >> 3;
      const bf16_t* wlds = mat ? sWv : sWf;
      v8f acc = {};
#pragma unroll
      for (int kt = 0; kt < 4; ++kt) {
        v16bf a  = load_A_tile(sP,   128, pixtile * 16, kt * 32, lane);
        v16bf bb = load_B_tile(wlds, 128, octile  * 16, kt * 32, lane);
        acc = WMMA_BF16(a, bb, acc);
      }
      int   ocl  = octile * 16 + (lane & 15);
      float bias = mat ? bvh[ocl] : bfh[ocl];
      bf16_t* dst = mat ? sPv : sPs;
#pragma unroll
      for (int rr = 0; rr < 8; ++rr) {
        int pix = pixtile * 16 + rr + ((lane >> 4) << 3);
        int n   = cp * 64 + pix;
        dst[n * 32 + ocl] = (bf16_t)(acc[rr] + bias);
      }
    }
    __syncthreads();
  }

  // ---- centers: adaptive 2x2 mean pool over 16x16 sub-quadrants ----
  if (t < 128) {
    int m = t >> 5, c = t & 31;
    int pw = m >> 1, ph = m & 1;
    float ss = 0.f, sv = 0.f;
    for (int x2 = 0; x2 < 16; ++x2)
      for (int y2 = 0; y2 < 16; ++y2) {
        int n = (pw * 16 + x2) * 32 + (ph * 16 + y2);
        ss += (float)sPs[n * 32 + c];
        sv += (float)sPv[n * 32 + c];
      }
    sCentS[t] = ss * (1.f / 256.f);
    sCentV[t] = sv * (1.f / 256.f);
  } else {
    sAgg[t - 128] = 0.f;
    if (t >= 252) sDen[t - 252] = 0.f;
  }
  __syncthreads();
  if (t < 4) {
    float nn = 0.f;
    for (int c = 0; c < 32; ++c) { float v = sCentS[t * 32 + c]; nn += v * v; }
    sNrm[t] = 1.f / fmaxf(sqrtf(nn), 1e-12f);
  }
  __syncthreads();
  if (t < 128) sCentS[t] *= sNrm[t >> 5];
  __syncthreads();

  // ---- per-point: cosine sim vs 4 centers, sigmoid, hard top-1, aggregate ----
#pragma unroll
  for (int i = 0; i < 4; ++i) {
    int n = i * 256 + t;
    float ps[32];
    float nn = 0.f;
#pragma unroll
    for (int c = 0; c < 32; ++c) { ps[c] = (float)sPs[n * 32 + c]; nn += ps[c] * ps[c]; }
    float inv = 1.f / fmaxf(sqrtf(nn), 1e-12f);
    float best = -3.4e38f; int bm = 0;
#pragma unroll
    for (int m = 0; m < 4; ++m) {
      float dot = 0.f;
#pragma unroll
      for (int c = 0; c < 32; ++c) dot += sCentS[m * 32 + c] * ps[c];
      float sim = 1.f / (1.f + __expf(-(beta + alpha * (dot * inv))));
      if (sim > best) { best = sim; bm = m; }   // strict > => first index on ties
    }
    sS[n] = best;
    sIdx[n] = (unsigned char)bm;
    atomicAdd(&sDen[bm], best);
#pragma unroll
    for (int c = 0; c < 32; ++c)
      atomicAdd(&sAgg[bm * 32 + c], best * (float)sPv[n * 32 + c]);
  }
  __syncthreads();
  if (t < 128) sAgg[t] = (sAgg[t] + sCentV[t]) / (sDen[t >> 5] + 1.0f);
  __syncthreads();

  // ---- dispatch back to points; pg is pixel-major -> 64B contiguous per point ----
#pragma unroll
  for (int i = 0; i < 4; ++i) {
    int n = i * 256 + t;
    float s = sS[n];
    int m = sIdx[n];
    int off = (f1 * 32 + (n >> 5)) * 64 + f2 * 32 + (n & 31);   // pixel in image
    size_t pbase = ((size_t)(b * 4096 + off)) * 256 + e * 32;
#pragma unroll
    for (int c = 0; c < 32; ++c)
      pg[pbase + c] = (bf16_t)(s * sAgg[m * 32 + c]);
  }
}

// Kernel 2: output projection, [128 px x 256 k] x [256 k x 128 oc] per block via WMMA.
__global__ __launch_bounds__(256) void out_conv_kernel(
    const bf16_t* __restrict__ pg, const float* __restrict__ Wp,
    const float* __restrict__ bp, float* __restrict__ out)
{
  __shared__ __align__(16) bf16_t sWp[128 * 256];   // 64 KB
  __shared__ __align__(16) bf16_t sX [128 * 256];   // 64 KB, [pix][k]
  const int t    = threadIdx.x;
  const int lane = t & 31;
  const int wvid = t >> 5;
  const int gp0  = blockIdx.x * 128;  // 128 flat pixels per block (never crosses batch)
  const int b    = gp0 >> 12;
  const int pix0 = gp0 & 4095;

#pragma unroll
  for (int i = 0; i < 128; ++i) {
    int idx = i * 256 + t;
    sWp[idx] = (bf16_t)Wp[idx];
  }
#ifdef ASYNC_LDS_OK
  // pg is pixel-major: 16B global segments map 1:1 onto sX[pix][ic] -> pure async copy
#pragma unroll
  for (int i = 0; i < 16; ++i) {
    int s   = i * 256 + t;
    int pix = s >> 5;
    int ic  = (s & 31) << 3;
    const bf16_t* src = pg + ((size_t)(b * 4096 + pix0 + pix)) * 256 + ic;
    async_copy16(src, &sX[pix * 256 + ic]);
  }
  wait_async0();
#else
#pragma unroll
  for (int i = 0; i < 16; ++i) {
    int s   = i * 256 + t;
    int pix = s >> 5;
    int ic  = (s & 31) << 3;
    const bf16_t* src = pg + ((size_t)(b * 4096 + pix0 + pix)) * 256 + ic;
    bf16_t* dst = &sX[pix * 256 + ic];
#pragma unroll
    for (int j = 0; j < 8; ++j) dst[j] = src[j];
  }
#endif
  __syncthreads();

  v8f acc[8] = {};
  const int pt = wvid;                 // each wave owns a 16-pixel row tile
#pragma unroll
  for (int kt = 0; kt < 8; ++kt) {
    v16bf a = load_A_tile(sX, 256, pt * 16, kt * 32, lane);
#pragma unroll
    for (int ot = 0; ot < 8; ++ot) {
      v16bf bb = load_B_tile(sWp, 256, ot * 16, kt * 32, lane);
      acc[ot] = WMMA_BF16(a, bb, acc[ot]);
    }
  }
#pragma unroll
  for (int ot = 0; ot < 8; ++ot) {
    int o = ot * 16 + (lane & 15);
    float bias = bp[o];
#pragma unroll
    for (int rr = 0; rr < 8; ++rr) {
      int pix = pt * 16 + rr + ((lane >> 4) << 3);
      out[(((size_t)(b * 128 + o)) << 12) + pix0 + pix] = acc[ot][rr] + bias;
    }
  }
}

extern "C" void kernel_launch(void* const* d_in, const int* in_sizes, int n_in,
                              void* d_out, int out_size, void* d_ws, size_t ws_size,
                              hipStream_t stream) {
  (void)in_sizes; (void)n_in; (void)out_size; (void)ws_size;
  const float* p   = (const float*)d_in[0];
  const float* Wf  = (const float*)d_in[1];
  const float* bf_ = (const float*)d_in[2];
  const float* Wv  = (const float*)d_in[3];
  const float* bv_ = (const float*)d_in[4];
  const float* Wp  = (const float*)d_in[5];
  const float* bp_ = (const float*)d_in[6];
  const float* al  = (const float*)d_in[7];
  const float* be  = (const float*)d_in[8];
  bf16_t* pg = (bf16_t*)d_ws;   // 32*4096*256 bf16 = 64 MiB scratch, pixel-major

  region_cluster_kernel<<<1024, 256, 0, stream>>>(p, Wf, bf_, Wv, bv_, al, be, pg);
  out_conv_kernel<<<1024, 256, 0, stream>>>(pg, Wp, bp_, (float*)d_out);
}
